// VideoEncoder_7541962572024
// MI455X (gfx1250) — compile-verified, matrix operand order checked by reading, not run on hardware
//
#include <hip/hip_runtime.h>
#include <hip/hip_bf16.h>
#include <stdint.h>

// ---------------------------------------------------------------- types
typedef __bf16 bf16;
typedef __bf16 bf16x16 __attribute__((ext_vector_type(16)));
typedef float  f32x8   __attribute__((ext_vector_type(8)));
typedef unsigned int u32x4 __attribute__((ext_vector_type(4)));
typedef int i32x4 __attribute__((ext_vector_type(4)));
typedef int i32x8 __attribute__((ext_vector_type(8)));

#define DEV __device__ __forceinline__

DEV float silu_f(float v) { return v / (1.f + __expf(-v)); }

// ---------------------------------------------------------------- WMMA fragment load
// LDS tile layout: [row][32] bf16 (row = M for A, N for B).
// 16-bit fragment layout (CDNA5 ISA 7.12.2): lanes 0-15 hold row, K in
// {0..7, 16..23}; lanes 16-31 hold K in {8..15, 24..31}; 2 bf16 per VGPR.
DEV bf16x16 load_frag(const bf16 (*tile)[32], int row0, int lane) {
    const uint32_t* p = (const uint32_t*)tile[row0 + (lane & 15)];
    const int o = (lane & 16) ? 4 : 0;
    union { bf16x16 v; uint32_t d[8]; } u;
#pragma unroll
    for (int i = 0; i < 4; i++) { u.d[i] = p[o + i]; u.d[4 + i] = p[o + 8 + i]; }
    return u.v;
}

// ---------------------------------------------------------------- TDM descriptor (D#) 2D tile load
// Loads a [rows x 32] bf16 tile (row stride = strideElems, in elements) from
// gptr into LDS at lds_off, row-major contiguous. Issued by thread 0 only (once per block).
DEV void tdm_load_tile(unsigned lds_off, const bf16* gptr, int rows, int strideElems) {
    const uint64_t ga = (uint64_t)(uintptr_t)gptr;
    u32x4 g0;
    g0[0] = 1u;                                               // count=1, user mode
    g0[1] = lds_off;                                          // lds_addr
    g0[2] = (unsigned)(ga & 0xffffffffu);                     // global_addr[31:0]
    g0[3] = (unsigned)((ga >> 32) & 0x1ffffffu) | (2u << 30); // addr[56:32] | type=2
    const unsigned td0 = (unsigned)strideElems;               // tensor_dim0
    const unsigned td1 = (unsigned)rows;                      // tensor_dim1
    const uint64_t st = (uint64_t)(unsigned)strideElems;      // tensor_dim0_stride
    i32x8 g1;
    g1[0] = (int)(1u << 16);                                  // data_size = 2B
    g1[1] = (int)((td0 & 0xffffu) << 16);                     // tensor_dim0[15:0]
    g1[2] = (int)(((td0 >> 16) & 0xffffu) | ((td1 & 0xffffu) << 16));
    g1[3] = (int)(((td1 >> 16) & 0xffffu) | (32u << 16));     // tile_dim0 = 32
    g1[4] = (int)(td1 & 0xffffu);                             // tile_dim1 = rows
    g1[5] = (int)(unsigned)(st & 0xffffffffu);
    g1[6] = (int)(unsigned)((st >> 32) & 0xffffu);
    g1[7] = 0;
    const i32x4 gz = {0, 0, 0, 0};
#if __clang_major__ >= 23
    const i32x8 gz8 = {0, 0, 0, 0, 0, 0, 0, 0};
    __builtin_amdgcn_tensor_load_to_lds(g0, g1, gz, gz, gz8, 0);
#else
    __builtin_amdgcn_tensor_load_to_lds(g0, g1, gz, gz, 0);
#endif
}

// ---------------------------------------------------------------- generic GEMM
// C[M,N](f32) = A[M,K](bf16 row-major) * Bt[N,K](bf16, pre-transposed)
// EP: 0 = store f32, 1 = C += (residual), 2 = silu -> bf16 Ob
// Double-buffered LDS; A and B tiles both fetched by the Tensor Data Mover.
template <int EP>
__global__ __launch_bounds__(256) void gemm_kernel(
    const bf16* __restrict__ A, const bf16* __restrict__ Bt,
    float* __restrict__ C, bf16* __restrict__ Ob, int M, int N, int K) {
    __shared__ __align__(16) bf16 As[2][64][32];
    __shared__ __align__(16) bf16 Bs[2][128][32];
    const int tid = threadIdx.x, lane = tid & 31, wid = tid >> 5;
    const int wm = wid >> 1, wn = wid & 1;
    const int m0 = blockIdx.y * 64, n0 = blockIdx.x * 128;

    f32x8 acc[4];
    const f32x8 zero = {0.f, 0.f, 0.f, 0.f, 0.f, 0.f, 0.f, 0.f};
#pragma unroll
    for (int s = 0; s < 4; s++) acc[s] = zero;

    const int nt = K / 32;

    // prologue: tile 0 via TDM (A and B), single tensorcnt drain + barrier
    if (tid == 0) {
        tdm_load_tile((unsigned)(size_t)&As[0][0][0], A + (size_t)m0 * K, 64, K);
        tdm_load_tile((unsigned)(size_t)&Bs[0][0][0], Bt + (size_t)n0 * K, 128, K);
        __builtin_amdgcn_s_wait_tensorcnt(0);
    }
    __syncthreads();

    for (int it = 0; it < nt; ++it) {
        const int cur = it & 1, nxt = cur ^ 1;
        if (it + 1 < nt && tid == 0) {
            const int k1 = (it + 1) * 32;
            tdm_load_tile((unsigned)(size_t)&As[nxt][0][0], A + (size_t)m0 * K + k1, 64, K);
            tdm_load_tile((unsigned)(size_t)&Bs[nxt][0][0], Bt + (size_t)n0 * K + k1, 128, K);
        }
        bf16x16 a = load_frag(As[cur], wm * 16, lane);
#pragma unroll
        for (int s = 0; s < 4; s++) {
            bf16x16 b = load_frag(Bs[cur], wn * 64 + s * 16, lane);
            acc[s] = __builtin_amdgcn_wmma_f32_16x16x32_bf16(
                false, a, false, b, (short)0, acc[s], false, false);
        }
        if (tid == 0) __builtin_amdgcn_s_wait_tensorcnt(0);
        __syncthreads();
    }

    const int mrow = m0 + wm * 16 + ((lane & 16) ? 8 : 0);
    const int ncol = n0 + wn * 64 + (lane & 15);
#pragma unroll
    for (int s = 0; s < 4; s++) {
#pragma unroll
        for (int r = 0; r < 8; r++) {
            const int m = mrow + r, n = ncol + s * 16;
            const float v = acc[s][r];
            if (EP == 0)      C[(size_t)m * N + n] = v;
            else if (EP == 1) C[(size_t)m * N + n] += v;
            else              Ob[(size_t)m * N + n] = (bf16)silu_f(v);
        }
    }
}

// ---------------------------------------------------------------- implicit-GEMM 3x3/s2/p1 conv + BN + SiLU
// Permuted GEMM-K order: k' = (kh*3+kw)*Ci + ci  (Ci is a power of two, %32==0)
// => every 32-wide K slice has constant (kh,kw): no per-element decode ALU.
// X: [N][Ci][Hi][Wi] bf16, W: [Co][9*Ci] bf16 (permuted), Y: [N][Co][Ho][Wo] bf16
__global__ __launch_bounds__(256) void conv_gemm_kernel(
    const bf16* __restrict__ X, const bf16* __restrict__ W, bf16* __restrict__ Y,
    const float* __restrict__ sc, const float* __restrict__ sh,
    int Ci, int ciLog2, int Hi, int Wi, int Co, int Ho, int Wo) {
    const int K = Ci * 9, HoWo = Ho * Wo;
    const size_t HiWi = (size_t)Hi * Wi;
    __shared__ __align__(16) bf16 As[2][64][32];
    __shared__ __align__(16) bf16 Bs[2][128][32];
    const int tid = threadIdx.x, lane = tid & 31, wid = tid >> 5;
    const int wm = wid >> 1, wn = wid & 1;
    const int m0 = blockIdx.y * 64, n0 = blockIdx.x * 128;

    f32x8 acc[4];
    const f32x8 zero = {0.f, 0.f, 0.f, 0.f, 0.f, 0.f, 0.f, 0.f};
#pragma unroll
    for (int s = 0; s < 4; s++) acc[s] = zero;

    // m-side im2col decode hoisted out of the K loop (one m per thread)
    const int am = tid >> 2, akb = (tid & 3) * 8;
    const int gm = m0 + am;
    const int img = gm / HoWo, rem = gm % HoWo;
    const int oh = rem / Wo, ow = rem % Wo;
    const int ihb = oh * 2 - 1, iwb = ow * 2 - 1;
    const size_t xbase = (size_t)img * Ci * HiWi;

    union { bf16 h[8]; uint4 v; } ar;
    const int nt = K / 32;

    auto load_a = [&](int k0) {
        const int r9 = k0 >> ciLog2;          // uniform (kh,kw) for whole slice
        const int cb_ = (k0 & (Ci - 1)) + akb;
        const int kh = r9 / 3, kw = r9 - kh * 3;
        const int ih = ihb + kh, iw = iwb + kw;
        if (ih >= 0 && ih < Hi && iw >= 0 && iw < Wi) {
            const bf16* src = X + xbase + (size_t)cb_ * HiWi + (size_t)ih * Wi + iw;
#pragma unroll
            for (int j = 0; j < 8; j++) ar.h[j] = src[j * HiWi];
        } else {
#pragma unroll
            for (int j = 0; j < 8; j++) ar.h[j] = (bf16)0.f;
        }
    };

    load_a(0);
    if (tid == 0)
        tdm_load_tile((unsigned)(size_t)&Bs[0][0][0], W + (size_t)n0 * K, 128, K);
    *(uint4*)&As[0][am][akb] = ar.v;
    if (tid == 0) __builtin_amdgcn_s_wait_tensorcnt(0);
    __syncthreads();

    for (int it = 0; it < nt; ++it) {
        const int cur = it & 1, nxt = cur ^ 1;
        if (it + 1 < nt) {
            const int k1 = (it + 1) * 32;
            load_a(k1);
            if (tid == 0)
                tdm_load_tile((unsigned)(size_t)&Bs[nxt][0][0],
                              W + (size_t)n0 * K + k1, 128, K);
            *(uint4*)&As[nxt][am][akb] = ar.v;
        }
        bf16x16 a = load_frag(As[cur], wm * 16, lane);
#pragma unroll
        for (int s = 0; s < 4; s++) {
            bf16x16 b = load_frag(Bs[cur], wn * 64 + s * 16, lane);
            acc[s] = __builtin_amdgcn_wmma_f32_16x16x32_bf16(
                false, a, false, b, (short)0, acc[s], false, false);
        }
        if (tid == 0) __builtin_amdgcn_s_wait_tensorcnt(0);
        __syncthreads();
    }

    const int mrow = m0 + wm * 16 + ((lane & 16) ? 8 : 0);
    const int ncol = n0 + wn * 64 + (lane & 15);
#pragma unroll
    for (int s = 0; s < 4; s++) {
#pragma unroll
        for (int r = 0; r < 8; r++) {
            const int m = mrow + r, co = ncol + s * 16;
            const int oimg = m / HoWo, orem = m % HoWo;
            float v = acc[s][r] * sc[co] + sh[co];
            Y[((size_t)oimg * Co + co) * HoWo + orem] = (bf16)silu_f(v);
        }
    }
}

// ---------------------------------------------------------------- conv1 (3->64, k7 s2 p3) VALU
__global__ __launch_bounds__(256) void conv1_kernel(
    const float* __restrict__ X, const float* __restrict__ Wt,
    const float* __restrict__ sc, const float* __restrict__ sh,
    bf16* __restrict__ Y) {
    __shared__ float wl[8 * 147];
    const int tid = threadIdx.x, coB = blockIdx.y * 8, img = blockIdx.z;
    for (int i = tid; i < 8 * 147; i += 256) wl[i] = Wt[(size_t)coB * 147 + i];
    __syncthreads();
    const int p = blockIdx.x * 256 + tid;  // 0..12543
    const int oh = p / 112, ow = p % 112;
    float acc[8];
#pragma unroll
    for (int c = 0; c < 8; c++) acc[c] = 0.f;
    for (int ci = 0; ci < 3; ci++)
        for (int kh = 0; kh < 7; kh++) {
            const int ih = oh * 2 - 3 + kh;
            if (ih < 0 || ih >= 224) continue;
            for (int kw = 0; kw < 7; kw++) {
                const int iw = ow * 2 - 3 + kw;
                if (iw < 0 || iw >= 224) continue;
                const float xv = X[(((size_t)img * 3 + ci) * 224 + ih) * 224 + iw];
                const int wo_ = ci * 49 + kh * 7 + kw;
#pragma unroll
                for (int c = 0; c < 8; c++) acc[c] += xv * wl[c * 147 + wo_];
            }
        }
#pragma unroll
    for (int c = 0; c < 8; c++) {
        const int co = coB + c;
        float v = acc[c] * sc[co] + sh[co];
        Y[(((size_t)img * 64 + co) * 112 + oh) * 112 + ow] = (bf16)silu_f(v);
    }
}

// ---------------------------------------------------------------- small utility kernels
__global__ void bnfold_kernel(const float* g, const float* b, const float* m,
                              const float* v, float* sc, float* sh, int n) {
    int i = blockIdx.x * 256 + threadIdx.x;
    if (i < n) {
        float s = g[i] * rsqrtf(v[i] + 1e-5f);
        sc[i] = s;
        sh[i] = b[i] - m[i] * s;
    }
}

__global__ void f2b_kernel(const float* __restrict__ in, bf16* __restrict__ out, int n) {
    int i = blockIdx.x * 256 + threadIdx.x;
    if (i < n) out[i] = (bf16)in[i];
}

// conv weight: [Co][Ci][9] f32 -> [Co][9*Ci] bf16 with k' = r9*Ci + ci
__global__ void permw_kernel(const float* __restrict__ in, bf16* __restrict__ out,
                             int Ci, int total) {
    int idx = blockIdx.x * 256 + threadIdx.x;
    if (idx < total) {
        const int ci9 = Ci * 9;
        const int co = idx / ci9, rem = idx - co * ci9;
        const int r9 = rem / Ci, ci = rem - r9 * Ci;
        out[idx] = (bf16)in[(size_t)co * ci9 + ci * 9 + r9];
    }
}

// f32 [Kd][Nd] -> bf16 [Nd][Kd] (transpose via 32x32 LDS tile); grid (Nd/32, Kd/32), 256 thr
__global__ __launch_bounds__(256) void transpose_f2b_kernel(
    const float* __restrict__ in, bf16* __restrict__ out, int Kd, int Nd) {
    __shared__ float t[32][33];
    const int tx = threadIdx.x & 31, ty = threadIdx.x >> 5;
    const int k0 = blockIdx.y * 32, n0 = blockIdx.x * 32;
#pragma unroll
    for (int r = 0; r < 4; r++)
        t[ty + r * 8][tx] = in[(size_t)(k0 + ty + r * 8) * Nd + n0 + tx];
    __syncthreads();
#pragma unroll
    for (int r = 0; r < 4; r++)
        out[(size_t)(n0 + ty + r * 8) * Kd + k0 + tx] = (bf16)t[tx][ty + r * 8];
}

__global__ void pool_kernel(const bf16* __restrict__ Y, float* __restrict__ F) {
    int i = blockIdx.x * 256 + threadIdx.x;  // (n*1024 + c), 262144 total
    const uint2* p = (const uint2*)(Y + (size_t)i * 196);
    float s = 0.f;
    for (int j = 0; j < 49; j++) {
        const uint2 u = p[j];
        union { uint32_t w; bf16 h[2]; } a, b;
        a.w = u.x; b.w = u.y;
        s += (float)a.h[0] + (float)a.h[1] + (float)b.h[0] + (float)b.h[1];
    }
    F[i] = s * (1.f / 196.f);
}

__global__ __launch_bounds__(256) void rmsnorm_kernel(
    const float* __restrict__ X, const float* __restrict__ w,
    float* __restrict__ of, bf16* __restrict__ ob) {
    __shared__ float red[256];
    const int row = blockIdx.x, tid = threadIdx.x;
    const float4 v = ((const float4*)(X + (size_t)row * 1024))[tid];
    red[tid] = v.x * v.x + v.y * v.y + v.z * v.z + v.w * v.w;
    __syncthreads();
    for (int o = 128; o > 0; o >>= 1) {
        if (tid < o) red[tid] += red[tid + o];
        __syncthreads();
    }
    const float inv = rsqrtf(red[0] * (1.f / 1024.f) + 1e-6f);
    const float4 wv = ((const float4*)w)[tid];
    const float o0 = v.x * inv * wv.x, o1 = v.y * inv * wv.y;
    const float o2 = v.z * inv * wv.z, o3 = v.w * inv * wv.w;
    if (of) {
        float4 r; r.x = o0; r.y = o1; r.z = o2; r.w = o3;
        ((float4*)(of + (size_t)row * 1024))[tid] = r;
    }
    if (ob) {
        bf16* d = ob + (size_t)row * 1024 + tid * 4;
        d[0] = (bf16)o0; d[1] = (bf16)o1; d[2] = (bf16)o2; d[3] = (bf16)o3;
    }
}

// VQ: per (token, q) find nearest of 1024 codes (DQ=128) and copy it out
__global__ __launch_bounds__(256) void vq_kernel(
    const float* __restrict__ xn, const float* __restrict__ cb,
    float* __restrict__ qout) {
    const int token = blockIdx.y, q = blockIdx.x, tid = threadIdx.x;
    __shared__ float xv[128];
    __shared__ float bd[256];
    __shared__ int bi[256];
    if (tid < 128) xv[tid] = xn[(size_t)token * 1024 + q * 128 + tid];
    __syncthreads();
    float bestd = 3.4e38f; int besti = 0;
    for (int k = tid; k < 1024; k += 256) {
        const float* c = cb + ((size_t)q * 1024 + k) * 128;
        float d = 0.f;
#pragma unroll 4
        for (int i = 0; i < 128; i++) { const float t = xv[i] - c[i]; d += t * t; }
        if (d < bestd) { bestd = d; besti = k; }
    }
    bd[tid] = bestd; bi[tid] = besti;
    __syncthreads();
    for (int o = 128; o > 0; o >>= 1) {
        if (tid < o && bd[tid + o] < bd[tid]) { bd[tid] = bd[tid + o]; bi[tid] = bi[tid + o]; }
        __syncthreads();
    }
    const float* c = cb + ((size_t)q * 1024 + bi[0]) * 128;
    if (tid < 128) qout[(size_t)token * 1024 + q * 128 + tid] = c[tid];
}

// attention: grid (NH=8, B=4), 64 threads (one per query), T=64, HD=128
__global__ __launch_bounds__(64) void attn_kernel(
    const float* __restrict__ Qp, const float* __restrict__ Kp,
    const float* __restrict__ Vp, float* __restrict__ Op) {
    const int h = blockIdx.x, b = blockIdx.y, tq = threadIdx.x;
    __shared__ float s[64][64];
    const float scale = 0.088388347648318447f;  // 1/sqrt(128)
    const float* qr = Qp + ((size_t)(b * 64 + tq) * 1024) + h * 128;
    for (int tk = 0; tk < 64; tk++) {
        const float* kr = Kp + ((size_t)(b * 64 + tk) * 1024) + h * 128;
        float a = 0.f;
        for (int d = 0; d < 128; d++) a += qr[d] * kr[d];
        s[tq][tk] = a * scale;
    }
    float mx = -3.4e38f;
    for (int tk = 0; tk < 64; tk++) mx = fmaxf(mx, s[tq][tk]);
    float sum = 0.f;
    for (int tk = 0; tk < 64; tk++) { const float e = __expf(s[tq][tk] - mx); s[tq][tk] = e; sum += e; }
    const float inv = 1.f / sum;
    for (int d = 0; d < 128; d++) {
        float a = 0.f;
        for (int tk = 0; tk < 64; tk++)
            a += s[tq][tk] * Vp[((size_t)(b * 64 + tk) * 1024) + h * 128 + d];
        Op[((size_t)(b * 64 + tq) * 1024) + h * 128 + d] = a * inv;
    }
}

__global__ void add_kernel(const float* a, const float* b, float* o, int n) {
    int i = blockIdx.x * 256 + threadIdx.x;
    if (i < n) o[i] = a[i] + b[i];
}

// ---------------------------------------------------------------- host
extern "C" void kernel_launch(void* const* d_in, const int* in_sizes, int n_in,
                              void* d_out, int out_size, void* d_ws, size_t ws_size,
                              hipStream_t stream) {
    (void)in_sizes; (void)n_in; (void)out_size; (void)ws_size;
    const float* x = (const float*)d_in[0];
    const float *ck[4], *cg[4], *cbta[4], *cm[4], *cv[4];
    for (int i = 0; i < 4; i++) {
        ck[i] = (const float*)d_in[1 + 5 * i + 0];
        cg[i] = (const float*)d_in[1 + 5 * i + 1];
        cbta[i] = (const float*)d_in[1 + 5 * i + 2];
        cm[i] = (const float*)d_in[1 + 5 * i + 3];
        cv[i] = (const float*)d_in[1 + 5 * i + 4];
    }
    const float* input_norm = (const float*)d_in[21];
    const float* tcb = (const float*)d_in[22];
    const float* dcb = (const float*)d_in[23];
    const float* tnorm = (const float*)d_in[24];
    const float* dnorm = (const float*)d_in[25];

    char* ws = (char*)d_ws;
    size_t off = 0;
    auto alloc = [&](size_t bytes) -> void* {
        off = (off + 255) & ~(size_t)255;
        void* p = ws + off;
        off += bytes;
        return p;
    };

    bf16* y1 = (bf16*)alloc((size_t)256 * 64 * 112 * 112 * 2);
    bf16* y2 = (bf16*)alloc((size_t)256 * 128 * 56 * 56 * 2);
    bf16* y3 = (bf16*)alloc((size_t)256 * 256 * 28 * 28 * 2);
    bf16* y4 = (bf16*)alloc((size_t)256 * 1024 * 14 * 14 * 2);
    bf16* wc2b = (bf16*)alloc((size_t)128 * 576 * 2);
    bf16* wc3b = (bf16*)alloc((size_t)256 * 1152 * 2);
    bf16* wc4b = (bf16*)alloc((size_t)1024 * 2304 * 2);
    bf16* wtb = (bf16*)alloc((size_t)4 * 12582912 * 2);  // 4 tf blocks, [N][K] layout
    float *sc[4], *sh[4];
    const int CoA[4] = {64, 128, 256, 1024};
    for (int i = 0; i < 4; i++) {
        sc[i] = (float*)alloc(CoA[i] * 4);
        sh[i] = (float*)alloc(CoA[i] * 4);
    }
    float* feat = (float*)alloc(262144 * 4);
    float* xn = (float*)alloc(262144 * 4);
    float* qt = (float*)alloc(262144 * 4);
    float* qd = (float*)alloc(262144 * 4);
    float* xt = (float*)alloc(262144 * 4);
    float* xd = (float*)alloc(262144 * 4);
    float* qb = (float*)alloc(262144 * 4);
    float* kb = (float*)alloc(262144 * 4);
    float* vb = (float*)alloc(262144 * 4);
    float* ob = (float*)alloc(262144 * 4);
    bf16* an = (bf16*)alloc(262144 * 2);
    bf16* h1 = (bf16*)alloc((size_t)256 * 4096 * 2);

    // BN folds + weight precision conversion / permutation / transposition
    for (int i = 0; i < 4; i++)
        bnfold_kernel<<<(CoA[i] + 255) / 256, 256, 0, stream>>>(
            cg[i], cbta[i], cm[i], cv[i], sc[i], sh[i], CoA[i]);
    permw_kernel<<<(128 * 576 + 255) / 256, 256, 0, stream>>>(ck[1], wc2b, 64, 128 * 576);
    permw_kernel<<<(256 * 1152 + 255) / 256, 256, 0, stream>>>(ck[2], wc3b, 128, 256 * 1152);
    permw_kernel<<<(1024 * 2304 + 255) / 256, 256, 0, stream>>>(ck[3], wc4b, 256, 1024 * 2304);
    for (int si = 0; si < 2; si++)
        for (int li = 0; li < 2; li++) {
            const int base = 26 + si * 16 + li * 8;
            bf16* blk = wtb + (size_t)(si * 2 + li) * 12582912;
            for (int w = 0; w < 4; w++)  // wq,wk,wv,wo: [1024][1024] -> [N][K]
                transpose_f2b_kernel<<<dim3(32, 32), 256, 0, stream>>>(
                    (const float*)d_in[base + 1 + w], blk + (size_t)w * 1048576, 1024, 1024);
            transpose_f2b_kernel<<<dim3(128, 32), 256, 0, stream>>>(  // w1 [1024][4096]
                (const float*)d_in[base + 6], blk + 4194304, 1024, 4096);
            transpose_f2b_kernel<<<dim3(32, 128), 256, 0, stream>>>(  // w2 [4096][1024]
                (const float*)d_in[base + 7], blk + 8388608, 4096, 1024);
        }

    // conv stack (NCHW, bf16 activations)
    conv1_kernel<<<dim3(49, 8, 256), 256, 0, stream>>>(x, ck[0], sc[0], sh[0], y1);
    conv_gemm_kernel<<<dim3(1, 12544), 256, 0, stream>>>(
        y1, wc2b, y2, sc[1], sh[1], 64, 6, 112, 112, 128, 56, 56);
    conv_gemm_kernel<<<dim3(2, 3136), 256, 0, stream>>>(
        y2, wc3b, y3, sc[2], sh[2], 128, 7, 56, 56, 256, 28, 28);
    conv_gemm_kernel<<<dim3(8, 784), 256, 0, stream>>>(
        y3, wc4b, y4, sc[3], sh[3], 256, 8, 28, 28, 1024, 14, 14);

    // pool -> rmsnorm -> VQ -> out-norm
    pool_kernel<<<1024, 256, 0, stream>>>(y4, feat);
    rmsnorm_kernel<<<256, 256, 0, stream>>>(feat, input_norm, xn, nullptr);
    vq_kernel<<<dim3(8, 256), 256, 0, stream>>>(xn, tcb, qt);
    vq_kernel<<<dim3(8, 256), 256, 0, stream>>>(xn, dcb, qd);
    rmsnorm_kernel<<<256, 256, 0, stream>>>(qt, tnorm, xt, nullptr);
    rmsnorm_kernel<<<256, 256, 0, stream>>>(qd, dnorm, xd, nullptr);

    // transformer stacks (temporal on xt, depth on xd)
    float* xs[2] = {xt, xd};
    for (int si = 0; si < 2; si++) {
        float* xcur = xs[si];
        for (int li = 0; li < 2; li++) {
            const int base = 26 + si * 16 + li * 8;
            const float* n1 = (const float*)d_in[base + 0];
            const float* n2 = (const float*)d_in[base + 5];
            bf16* blk = wtb + (size_t)(si * 2 + li) * 12582912;
            bf16 *wqb = blk, *wkb = blk + 1048576, *wvb = blk + 2097152;
            bf16 *wob = blk + 3145728, *w1b = blk + 4194304, *w2b = blk + 8388608;

            rmsnorm_kernel<<<256, 256, 0, stream>>>(xcur, n1, nullptr, an);
            gemm_kernel<0><<<dim3(8, 4), 256, 0, stream>>>(an, wqb, qb, nullptr, 256, 1024, 1024);
            gemm_kernel<0><<<dim3(8, 4), 256, 0, stream>>>(an, wkb, kb, nullptr, 256, 1024, 1024);
            gemm_kernel<0><<<dim3(8, 4), 256, 0, stream>>>(an, wvb, vb, nullptr, 256, 1024, 1024);
            attn_kernel<<<dim3(8, 4), 64, 0, stream>>>(qb, kb, vb, ob);
            f2b_kernel<<<1024, 256, 0, stream>>>(ob, an, 262144);
            gemm_kernel<1><<<dim3(8, 4), 256, 0, stream>>>(an, wob, xcur, nullptr, 256, 1024, 1024);
            rmsnorm_kernel<<<256, 256, 0, stream>>>(xcur, n2, nullptr, an);
            gemm_kernel<2><<<dim3(32, 4), 256, 0, stream>>>(an, w1b, nullptr, h1, 256, 4096, 1024);
            gemm_kernel<1><<<dim3(8, 4), 256, 0, stream>>>(h1, w2b, xcur, nullptr, 256, 1024, 4096);
        }
    }

    add_kernel<<<1024, 256, 0, stream>>>(xt, xd, (float*)d_out, 262144);
}
